// EquiCtsConvBase_56805237457228
// MI455X (gfx1250) — compile-verified, matrix operand order checked by reading, not run on hardware
//
#include <hip/hip_runtime.h>
#include <hip/hip_bf16.h>
#include <math.h>

#define B_    2
#define M_    384
#define N_    384
#define CIN   8
#define COUT  8
#define NR    4      // W axis of sample grid (radius bins)
#define NTH   8      // theta bins (padded to 10 = H axis)
#define KDIM  640    // 40 taps * 16 (i,x)
#define SSTRIDE 644  // bucket row stride (pad for LDS bank spread)
#define NWAVES 8
#define KPW   (KDIM / NWAVES)   // 80 k-values per wave in stage 2

typedef __attribute__((ext_vector_type(2)))  float    v2f;
typedef __attribute__((ext_vector_type(8)))  float    v8f;
typedef __attribute__((ext_vector_type(16))) _Float16 v16h;

// Async global->LDS copy (CDNA5): per-lane, tracked by ASYNCcnt, bypasses VGPRs.
// LDS byte address = low 32 bits of the generic pointer (flat aperture rule).
__device__ __forceinline__ void async_copy_b32(unsigned lds_byte, unsigned gbyte_off,
                                               unsigned long long sbase) {
    asm volatile("global_load_async_to_lds_b32 %0, %1, %2"
                 :: "v"(lds_byte), "v"(gbyte_off), "s"(sbase)
                 : "memory");
}
__device__ __forceinline__ void wait_async0() {
    asm volatile("s_wait_asynccnt 0x0" ::: "memory");
}

__global__ __launch_bounds__(256)
void equi_cts_conv_kernel(const float* __restrict__ field,
                          const float* __restrict__ center,
                          const float* __restrict__ ffeat,
                          const float* __restrict__ fmask,
                          const float* __restrict__ kern,
                          float* __restrict__ out)
{
    __shared__ float sF[N_ * 16];          // f[n][(i,x)]            24 KB
    __shared__ float sKmat[KDIM * 16];     // Kmat[k=(tap,q)][p]     40 KB
    __shared__ float sS[16 * SSTRIDE];     // buckets S[m][k]        40 KB
    __shared__ float sOut[256];            // reduced 16x16 result
    __shared__ float sPsi[16];
    __shared__ float sFxy[N_ * 2];
    __shared__ float sCxy[32];
    __shared__ float sMask[N_];

    const int tid  = threadIdx.x;
    const int b    = blockIdx.x / (M_ / 16);
    const int m0   = (blockIdx.x % (M_ / 16)) * 16;

    // ---------------- stage 0: async DMA-style loads + Kmat build ----------------
    {
        unsigned long long fb = (unsigned long long)(const void*)(field + (size_t)b * N_ * 2);
        for (int i = tid; i < N_ * 2; i += 256)
            async_copy_b32((unsigned)(size_t)&sFxy[i], (unsigned)(i * 4), fb);
        unsigned long long ffb = (unsigned long long)(const void*)(ffeat + (size_t)b * N_ * 16);
        for (int i = tid; i < N_ * 16; i += 256)
            async_copy_b32((unsigned)(size_t)&sF[i], (unsigned)(i * 4), ffb);
        unsigned long long mb = (unsigned long long)(const void*)(fmask + (size_t)b * N_);
        for (int i = tid; i < N_; i += 256)
            async_copy_b32((unsigned)(size_t)&sMask[i], (unsigned)(i * 4), mb);
    }
    if (tid < 32) sCxy[tid] = center[(size_t)b * M_ * 2 + m0 * 2 + tid];

    __builtin_prefetch(kern, 0, 0);   // global_prefetch_b8
    // Kmat[k][p]: k = (iy*4+ix)*16 + (i*2+x), p = o*2+y, with theta wrap-padding
    for (int idx = tid; idx < KDIM * 16; idx += 256) {
        int p  = idx & 15;          // (o,y)
        int k  = idx >> 4;
        int q  = k & 15;            // (i,x)
        int c  = k >> 4;            // tap = iy*4+ix
        int ix = c & 3, iy = c >> 2;
        int it = (iy == 0) ? (NTH - 1) : ((iy == NTH + 1) ? 0 : iy - 1);
        int o  = p >> 1, y = p & 1;
        int i  = q >> 1, x = q & 1;
        sKmat[idx] = kern[((((o * CIN + i) * NR + ix) * NTH + it) * 2 + y) * 2 + x];
    }
    for (int i = tid; i < 16 * SSTRIDE; i += 256) sS[i] = 0.0f;
    for (int i = tid; i < 256; i += 256)          sOut[i] = 0.0f;
    if (tid < 16) sPsi[tid] = 0.0f;

    wait_async0();          // drain this wave's ASYNCcnt before the barrier
    __syncthreads();

    // ---------------- stage 1: geometry + bucket scatter ----------------
    // pidx = n*16 + ml  =>  ml = tid & 15 is loop-invariant per thread.
    const float invR = 1.0f / 1.5f;
    const int   ml   = tid & 15;
    const float cx   = sCxy[2 * ml];
    const float cy   = sCxy[2 * ml + 1];
    float*      srow = &sS[ml * SSTRIDE];
    float psi_local  = 0.0f;

    for (int pidx = tid; pidx < 16 * N_; pidx += 256) {
        int n = pidx >> 4;
        float relx = (sFxy[2 * n]     - cx) * invR;
        float rely = (sFxy[2 * n + 1] - cy) * invR;
        float rr = relx * relx + rely * rely;
        float r  = sqrtf(rr + 1e-9f);
        float theta = (relx == 0.0f && rely == 0.0f) ? 0.0f : atan2f(rely, relx);
        float gx = 2.0f * r - 1.0f;
        float gy = theta * 0.31830988618379f * ((float)NTH / (NTH + 2.0f));
        float ixf = (gx + 1.0f) * 2.0f - 0.5f;   // ((gx+1)*4 - 1)/2
        float iyf = (gy + 1.0f) * 5.0f - 0.5f;   // ((gy+1)*10 - 1)/2
        float fx0 = floorf(ixf), fy0 = floorf(iyf);
        float wx = ixf - fx0,  wy = iyf - fy0;
        int   x0 = (int)fx0,   y0 = (int)fy0;

        float t1  = 1.0f - rr;
        float att = (t1 > 0.0f) ? t1 * t1 * t1 : 0.0f;
        att *= sMask[n];
        psi_local += att;

        if (att != 0.0f) {
            const float* fv = &sF[n * 16];
            float wts[4] = { (1.0f - wx) * (1.0f - wy), wx * (1.0f - wy),
                             (1.0f - wx) * wy,          wx * wy };
            int xs[4] = { x0, x0 + 1, x0,     x0 + 1 };
            int ys[4] = { y0, y0,     y0 + 1, y0 + 1 };
            #pragma unroll
            for (int t = 0; t < 4; ++t) {
                int xx = xs[t], yy = ys[t];
                if (xx >= 0 && xx < NR && yy >= 0 && yy < NTH + 2) {
                    float cw = wts[t] * att;
                    float* sp = srow + (yy * NR + xx) * 16;
                    #pragma unroll
                    for (int q = 0; q < 16; ++q)
                        atomicAdd(&sp[q], cw * fv[q]);   // ds_add_f32
                }
            }
        }
    }
    atomicAdd(&sPsi[ml], psi_local);   // one atomic per thread
    __syncthreads();

    // ------ stage 2: Out(16x16) = Kmat(16x640) x S^T(640x16), K split over 8 waves ------
    {
        const int lane = tid & 31;
        const int wave = tid >> 5;          // 0..7, owns k in [wave*80, wave*80+80)
        const int h    = lane >> 4;         // lane half
        const int p    = lane & 15;         // A row (o,y) / B col (m)
        v8f acc = {};
#if defined(__has_builtin) && __has_builtin(__builtin_amdgcn_wmma_f32_16x16x4_f32)
        const int k0 = wave * KPW;
        for (int kb = k0; kb < k0 + KPW; kb += 4) {
            int k = kb + 2 * h;             // A/B: K = vgpr + 2*half
            v2f a, bv;
            a.x  = sKmat[k * 16 + p];
            a.y  = sKmat[(k + 1) * 16 + p];
            bv.x = sS[p * SSTRIDE + k];
            bv.y = sS[p * SSTRIDE + k + 1];
            acc = __builtin_amdgcn_wmma_f32_16x16x4_f32(
                false, a, false, bv, (short)0, acc, false, false);
        }
#else   // fallback: f16 WMMA, K chunks of 32
        const int k0 = wave * KPW;
        for (int kb = k0; kb < k0 + KPW; kb += 32) {
            v16h a, bv;
            #pragma unroll
            for (int e = 0; e < 16; ++e) {
                int k = kb + ((e >> 3) * 16) + h * 8 + (e & 7);
                a[e]  = (_Float16)sKmat[k * 16 + p];
                bv[e] = (_Float16)sS[p * SSTRIDE + k];
            }
            acc = __builtin_amdgcn_wmma_f32_16x16x32_f16(
                false, a, false, bv, (short)0, acc, false, false);
        }
#endif
        // reduce partial accumulators across waves: D row = vgpr + 8*half, col = lane&15
        #pragma unroll
        for (int j = 0; j < 8; ++j) {
            int row = j + 8 * h;
            atomicAdd(&sOut[row * 16 + p], acc[j]);
        }
    }
    __syncthreads();

    // ---------------- stage 3: psi-normalize and store (all 256 threads) ----------------
    {
        int col = tid & 15;          // m within tile
        int row = tid >> 4;          // (o,y)
        float psi = sPsi[col];
        if (psi == 0.0f) psi = 1.0f;
        out[((size_t)(b * M_ + m0 + col)) * 16 + row] = sOut[row * 16 + col] / psi;
    }
}

extern "C" void kernel_launch(void* const* d_in, const int* in_sizes, int n_in,
                              void* d_out, int out_size, void* d_ws, size_t ws_size,
                              hipStream_t stream) {
    (void)in_sizes; (void)n_in; (void)out_size; (void)d_ws; (void)ws_size;
    const float* field  = (const float*)d_in[0];
    const float* center = (const float*)d_in[1];
    const float* ffeat  = (const float*)d_in[2];
    const float* fmask  = (const float*)d_in[3];
    const float* kern   = (const float*)d_in[4];
    float* out = (float*)d_out;

    dim3 grid(B_ * (M_ / 16));   // 48 workgroups
    dim3 block(256);             // 8 wave32 waves
    equi_cts_conv_kernel<<<grid, block, 0, stream>>>(field, center, ffeat, fmask, kern, out);
}
